// ModulatedDeformConvPack_52553219834668
// MI455X (gfx1250) — matrix operand order repaired and checked by reading, not compiled, not run on hardware
//
#include <hip/hip_runtime.h>
#include <hip/hip_bf16.h>

// ---------------------------------------------------------------------------
// Modulated deformable conv (DCNv2) for gfx1250 / MI455X.
//  Kernel 1: 27-ch offset conv as fp32 WMMA GEMM (v_wmma_f32_16x16x4_f32)
//            M=27(pad 32) K=576 N=B*H*W ; raw off_full(+bias) -> workspace
//  Kernel 2: fused bilinear-sample im2col (bf16, LDS) + bf16 WMMA GEMM
//            (v_wmma_f32_16x16x32_bf16): out[o,n] = sum_ck W[o,ck]*V[ck,n]
// ---------------------------------------------------------------------------

typedef __attribute__((ext_vector_type(16))) __bf16 v16bf;
typedef __attribute__((ext_vector_type(8)))  __bf16 v8bf;
typedef __attribute__((ext_vector_type(8)))  float  v8f;
typedef __attribute__((ext_vector_type(2)))  float  v2f;

#define CDIM   64
#define ODIM   64
#define KKPT   9          // 3x3 kernel points
#define KD     576        // CDIM * KKPT  (GEMM K)
#define HW     16384      // 128*128
#define WIDTH  128
#define HEIGHT 128
#define BATCH  4
#define NOFF   27         // 3*KKPT offset-conv channels

static __device__ inline int iclampi(int v, int lo, int hi) {
  return v < lo ? lo : (v > hi ? hi : v);
}

// ---------------------------------------------------------------------------
// Kernel 1: offset conv (3x3, pad 1) as fp32 WMMA GEMM.
// Block = 64 threads (2 waves); tile = 16 pixels x 32 rows (27 valid).
// Wave w owns rows [16w, 16w+16). im2col panel (fp32) staged in LDS.
// off_buf layout: ((b*27 + ch) << 14) + hw   (raw off_full + bias)
// ---------------------------------------------------------------------------
#define T1N    16
#define KPAD1  584   // KD + 8 pad; row byte-stride 2336 (8-aligned v2f runs)

__global__ __launch_bounds__(64)
void dcn_offset_kernel(const float* __restrict__ x,
                       const float* __restrict__ ow,
                       const float* __restrict__ ob,
                       float* __restrict__ off_buf) {
  __shared__ __attribute__((aligned(16))) float Plds[T1N * KPAD1];

  const int tid  = threadIdx.x;
  const int lane = tid & 31;
  const int wv   = tid >> 5;     // wave id 0..1 -> M tile
  const int half = lane >> 4;
  const int l16  = lane & 15;
  const int n0   = blockIdx.x * T1N;

  // ---- Phase 1: zero-padded im2col panel -> Plds[pixel][ck] (fp32) ----
  {
    const int p  = tid >> 2;            // pixel 0..15
    const int c0 = (tid & 3) * 16;      // channel sub-range
    const int n  = n0 + p;
    const int b  = n >> 14;
    const int hw = n & (HW - 1);
    const int h  = hw >> 7;
    const int w  = hw & (WIDTH - 1);
    for (int c = c0; c < c0 + 16; ++c) {
      const float* xc = x + (((b * CDIM + c) << 14));
      #pragma unroll
      for (int t = 0; t < KKPT; ++t) {
        const int yy = h - 1 + (t / 3);
        const int xx = w - 1 + (t % 3);
        const bool okv = (yy >= 0) && (yy < HEIGHT) && (xx >= 0) && (xx < WIDTH);
        Plds[p * KPAD1 + c * KKPT + t] = okv ? xc[yy * WIDTH + xx] : 0.0f;
      }
    }
  }
  __syncthreads();

  // ---- Phase 2: GEMM with v_wmma_f32_16x16x4_f32, K = 144 x 4 ----
  const int row = wv * 16 + l16;                    // A row (27..31 padded)
  const float zf = (row < NOFF) ? 1.0f : 0.0f;
  const float* arow = ow + (row < NOFF ? row : 0) * KD;
  __builtin_prefetch(arow, 0, 1);                   // global_prefetch_b8

  v8f acc = {};
  #pragma unroll
  for (int kc = 0; kc < KD / 4; ++kc) {
    const int kb = kc * 4 + 2 * half;               // K = 2*half + j
    v2f a = *(const v2f*)(arow + kb);
    a *= zf;                                        // zero padded rows
    const v2f bfr = *(const v2f*)(&Plds[l16 * KPAD1 + kb]);
    acc = __builtin_amdgcn_wmma_f32_16x16x4_f32(
        false, a, false, bfr, (short)0, acc, false, false);
  }

  // ---- Epilogue: bias + store raw off_full (rows >= 27 dropped) ----
  {
    const int p  = l16;
    const int n  = n0 + p;
    const int b  = n >> 14;
    const int hw = n & (HW - 1);
    float* obase = off_buf + ((b * NOFF) << 14) + hw;
    #pragma unroll
    for (int r = 0; r < 8; ++r) {
      const int o = wv * 16 + 8 * half + r;         // D row = r + 8*half
      if (o < NOFF) obase[o << 14] = acc[r] + ob[o];
    }
  }
}

// ---------------------------------------------------------------------------
// Kernel 2: fused bilinear sampling + bf16 WMMA GEMM.
// Block = 128 threads (4 waves). Tile = 32 pixels x 64 output channels.
// Wave w owns output rows [16w, 16w+16); two 16-col WMMA accumulators.
// ---------------------------------------------------------------------------
#define TILE_N 32
#define KPAD   584   // KD + 8 pad: row byte-stride 1168 = 73*16 (b128-aligned)

__global__ __launch_bounds__(128)
void dcn_main_kernel(const float* __restrict__ x,
                     const float* __restrict__ off_buf,
                     const float* __restrict__ dw,
                     const float* __restrict__ db,
                     float* __restrict__ out) {
  __shared__ __attribute__((aligned(16))) __bf16 Blds[TILE_N * KPAD];

  const int tid  = threadIdx.x;
  const int lane = tid & 31;
  const int wv   = tid >> 5;     // wave id 0..3 -> M tile
  const int half = lane >> 4;
  const int l16  = lane & 15;
  const int n0   = blockIdx.x * TILE_N;

  __builtin_prefetch(dw + (wv * 16 + l16) * KD, 0, 1);

  // ---- Phase 1: sampled/masked im2col panel -> Blds[pixel][ck] (bf16) ----
  {
    const int p  = tid >> 2;            // pixel 0..31
    const int c0 = (tid & 3) * 16;      // channel sub-range
    const int n  = n0 + p;
    const int b  = n >> 14;
    const int hw = n & (HW - 1);
    const int h  = hw >> 7;
    const int w  = hw & (WIDTH - 1);
    const float* xb  = x + ((b * CDIM) << 14);
    const float* ofb = off_buf + ((b * NOFF) << 14);

    #pragma unroll
    for (int k = 0; k < KKPT; ++k) {
      // offset[c] = off_full[c < 9 ? c : c + 9]; dy_k=offset[2k], dx_k=offset[2k+1]
      const int cy  = 2 * k;
      const int cx  = 2 * k + 1;
      const int chy = (cy < KKPT) ? cy : cy + KKPT;
      const int chx = (cx < KKPT) ? cx : cx + KKPT;
      const float dy = ofb[(chy << 14) + hw];
      const float dx = ofb[(chx << 14) + hw];
      const float m  = 1.0f / (1.0f + __expf(-ofb[((KKPT + k) << 14) + hw]));
      const float py = (float)(h - 1 + (k / 3)) + dy;
      const float px = (float)(w - 1 + (k % 3)) + dx;
      const float fy = floorf(py);
      const float fx = floorf(px);
      const float ly = py - fy, lx = px - fx;
      const int y0 = (int)fy, x0 = (int)fx;
      const int y1 = y0 + 1,  x1 = x0 + 1;
      const float vy0 = (y0 >= 0 && y0 < HEIGHT) ? 1.f : 0.f;
      const float vy1 = (y1 >= 0 && y1 < HEIGHT) ? 1.f : 0.f;
      const float vx0 = (x0 >= 0 && x0 < WIDTH)  ? 1.f : 0.f;
      const float vx1 = (x1 >= 0 && x1 < WIDTH)  ? 1.f : 0.f;
      const int i00 = iclampi(y0, 0, HEIGHT - 1) * WIDTH + iclampi(x0, 0, WIDTH - 1);
      const int i01 = iclampi(y0, 0, HEIGHT - 1) * WIDTH + iclampi(x1, 0, WIDTH - 1);
      const int i10 = iclampi(y1, 0, HEIGHT - 1) * WIDTH + iclampi(x0, 0, WIDTH - 1);
      const int i11 = iclampi(y1, 0, HEIGHT - 1) * WIDTH + iclampi(x1, 0, WIDTH - 1);
      const float w00 = (1.f - ly) * (1.f - lx) * vy0 * vx0 * m;
      const float w01 = (1.f - ly) * lx         * vy0 * vx1 * m;
      const float w10 = ly * (1.f - lx)         * vy1 * vx0 * m;
      const float w11 = ly * lx                 * vy1 * vx1 * m;
      for (int c = 0; c < 16; ++c) {
        const float* xc = xb + ((c0 + c) << 14);
        const float val = xc[i00] * w00 + xc[i01] * w01 +
                          xc[i10] * w10 + xc[i11] * w11;
        Blds[p * KPAD + (c0 + c) * KKPT + k] = (__bf16)val;
      }
    }
  }
  __syncthreads();

  // ---- Phase 2: GEMM with v_wmma_f32_16x16x32_bf16, K = 18 x 32 ----
  v8f acc[2] = {v8f{}, v8f{}};
  const int    row  = wv * 16 + l16;          // A row for this lane
  const float* wrow = dw + row * KD;

  #pragma unroll
  for (int kc = 0; kc < 18; ++kc) {
    // A fragment (ISA 16-bit A 16x32 layout): runs K = kc*32+8h+[0,8) and +16
    const int k1 = kc * 32 + 8 * half;
    const float4 a0 = *(const float4*)(wrow + k1);
    const float4 a1 = *(const float4*)(wrow + k1 + 4);
    const float4 a2 = *(const float4*)(wrow + k1 + 16);
    const float4 a3 = *(const float4*)(wrow + k1 + 20);
    v16bf afrag;
    afrag[0]  = (__bf16)a0.x; afrag[1]  = (__bf16)a0.y;
    afrag[2]  = (__bf16)a0.z; afrag[3]  = (__bf16)a0.w;
    afrag[4]  = (__bf16)a1.x; afrag[5]  = (__bf16)a1.y;
    afrag[6]  = (__bf16)a1.z; afrag[7]  = (__bf16)a1.w;
    afrag[8]  = (__bf16)a2.x; afrag[9]  = (__bf16)a2.y;
    afrag[10] = (__bf16)a2.z; afrag[11] = (__bf16)a2.w;
    afrag[12] = (__bf16)a3.x; afrag[13] = (__bf16)a3.y;
    afrag[14] = (__bf16)a3.z; afrag[15] = (__bf16)a3.w;

    // B fragments (K runs 8*half+[0,8) and +16, N = l16): matches A layout.
    const __bf16* bp0 = &Blds[(0 * 16 + l16) * KPAD + kc * 32 + 8 * half];
    const __bf16* bp1 = &Blds[(1 * 16 + l16) * KPAD + kc * 32 + 8 * half];
    const v8bf b0lo = *(const v8bf*)bp0;
    const v8bf b0hi = *(const v8bf*)(bp0 + 16);
    const v8bf b1lo = *(const v8bf*)bp1;
    const v8bf b1hi = *(const v8bf*)(bp1 + 16);
    const v16bf bf0 = __builtin_shufflevector(b0lo, b0hi,
        0,1,2,3,4,5,6,7,8,9,10,11,12,13,14,15);
    const v16bf bf1 = __builtin_shufflevector(b1lo, b1hi,
        0,1,2,3,4,5,6,7,8,9,10,11,12,13,14,15);

    acc[0] = __builtin_amdgcn_wmma_f32_16x16x32_bf16(
        false, afrag, false, bf0, (short)0, acc[0], false, false);
    acc[1] = __builtin_amdgcn_wmma_f32_16x16x32_bf16(
        false, afrag, false, bf1, (short)0, acc[1], false, false);
  }

  // ---- Epilogue: bias + scatter store (D layout: row = r + 8*half) ----
  #pragma unroll
  for (int nt = 0; nt < 2; ++nt) {
    const int p  = nt * 16 + l16;
    const int n  = n0 + p;
    const int b  = n >> 14;
    const int hw = n & (HW - 1);
    float* obp = out + ((b * ODIM) << 14) + hw;
    #pragma unroll
    for (int r = 0; r < 8; ++r) {
      const int o = wv * 16 + 8 * half + r;
      obp[o << 14] = acc[nt][r] + db[o];
    }
  }
}

// ---------------------------------------------------------------------------
extern "C" void kernel_launch(void* const* d_in, const int* in_sizes, int n_in,
                              void* d_out, int out_size, void* d_ws, size_t ws_size,
                              hipStream_t stream) {
  const float* x  = (const float*)d_in[0];   // (4, 64, 128, 128)
  const float* ow = (const float*)d_in[1];   // (27, 64, 3, 3)
  const float* ob = (const float*)d_in[2];   // (27,)
  const float* dw = (const float*)d_in[3];   // (64, 64, 3, 3)
  const float* db = (const float*)d_in[4];   // (64,)
  float* out      = (float*)d_out;           // (4, 64, 128, 128)
  float* off_buf  = (float*)d_ws;            // 4*27*16384 floats = 6.75 MB

  dcn_offset_kernel<<<(BATCH * HW) / T1N, 64, 0, stream>>>(x, ow, ob, off_buf);
  dcn_main_kernel<<<(BATCH * HW) / TILE_N, 128, 0, stream>>>(x, off_buf, dw, db, out);
}
